// AdaptiveGravityAttention_90649579749474
// MI455X (gfx1250) — matrix operand order repaired and checked by reading, not compiled, not run on hardware
//
#include <hip/hip_runtime.h>

// ---------------------------------------------------------------------------
// AdaptiveGravityAttention for MI455X (gfx1250): bf16 WMMA everywhere heavy.
// B=4, T=2048, C=1024, H=16, D=64.
// ---------------------------------------------------------------------------

#define BB 4
#define TT 2048
#define CC 1024
#define HH 16
#define DD 64
#define MM (BB * TT)      // 8192 rows
#define N3 (3 * CC)       // 3072

typedef __attribute__((ext_vector_type(16))) __bf16 v16bf;
typedef __attribute__((ext_vector_type(8)))  float  v8f;
typedef __attribute__((ext_vector_type(4)))  unsigned int u32x4;

union FragAB { v16bf v; u32x4 u[2]; };

// A-matrix (16xK) per-lane layout: elem i -> K = k0 + 16*g + i
// (one contiguous 32-byte run per lane).
static __device__ inline v16bf frag_rowA(const __bf16* row, int k0, int g) {
  FragAB f;
  f.u[0] = *(const u32x4*)(row + k0 + 16 * g);
  f.u[1] = *(const u32x4*)(row + k0 + 16 * g + 8);
  return f.v;
}
// B-matrix (Kx16) per-lane layout via Bt[n][k] row-major: elem i -> K = k0 + 16*g + i
// => one contiguous 32-byte run per lane.
static __device__ inline v16bf frag_rowB(const __bf16* row, int k0, int g) {
  const u32x4* p = (const u32x4*)(row + k0 + 16 * g);
  FragAB f;
  f.u[0] = p[0];
  f.u[1] = p[1];
  return f.v;
}

static __device__ inline v8f zero8() {
  v8f z;
#pragma unroll
  for (int i = 0; i < 8; ++i) z[i] = 0.f;
  return z;
}

#define WMMA_BF16(a, b, c) \
  __builtin_amdgcn_wmma_f32_16x16x32_bf16(false, (a), false, (b), (short)0, (c), false, false)

// ---------------------------------------------------------------------------
// fp32 -> bf16 elementwise
__global__ __launch_bounds__(256) void cvt_x_kernel(const float* __restrict__ x,
                                                    __bf16* __restrict__ xb, int n) {
  int i = blockIdx.x * 256 + threadIdx.x;
  if (i < n) xb[i] = (__bf16)x[i];
}

// fp32 [K,N] -> bf16 transposed Bt [N,K]
__global__ __launch_bounds__(256) void cvt_wt_kernel(const float* __restrict__ w,
                                                     __bf16* __restrict__ wt,
                                                     int K, int N) {
  int i = blockIdx.x * 256 + threadIdx.x;
  if (i < K * N) {
    int n = i / K;
    int k = i - n * K;
    wt[i] = (__bf16)w[(size_t)k * N + n];
  }
}

// ---------------------------------------------------------------------------
// gval[row][h] = softplus(x[row,:] @ g_w[:,h] + g_b[h]),  row = b*T + t
__global__ __launch_bounds__(256) void gravity_kernel(const float* __restrict__ x,
                                                      const float* __restrict__ g_w,
                                                      const float* __restrict__ g_b,
                                                      float* __restrict__ gval) {
  int row = blockIdx.x;
  int tid = threadIdx.x;
  __shared__ float s[HH];
  if (tid < HH) s[tid] = 0.f;
  __syncthreads();
  float acc[HH];
#pragma unroll
  for (int h = 0; h < HH; ++h) acc[h] = 0.f;
  for (int c = tid; c < CC; c += 256) {
    float xv = x[(size_t)row * CC + c];
#pragma unroll
    for (int h = 0; h < HH; ++h) acc[h] += xv * g_w[c * HH + h];
  }
#pragma unroll
  for (int h = 0; h < HH; ++h) atomicAdd(&s[h], acc[h]);
  __syncthreads();
  if (tid < HH) {
    float z = s[tid] + g_b[tid];
    float sp = (z > 20.f) ? z : log1pf(__expf(z));
    gval[(size_t)row * HH + tid] = sp;
  }
}

// ---------------------------------------------------------------------------
// qkv = xb @ W_attn (bf16 WMMA, f32 acc) with fused RoPE + head-reorg epilogue.
// One wave computes a 16(M) x 64(N) tile; N-tile == one head's D for q/k/v.
// q -> qh[B,H,T,D] (rope'd, bf16), k -> kh (rope'd), v -> vt[B,H,D,T] (transposed).
__global__ __launch_bounds__(128) void qkv_gemm_rope_kernel(
    const __bf16* __restrict__ xb, const __bf16* __restrict__ wbt,
    const float* __restrict__ rope, __bf16* __restrict__ qh,
    __bf16* __restrict__ kh, __bf16* __restrict__ vt) {
  const int lane = threadIdx.x & 31;
  const int w = blockIdx.x * 4 + (threadIdx.x >> 5);  // 0 .. 24575
  const int mt = w % (MM / 16);                       // 0 .. 511
  const int nt = w / (MM / 16);                       // 0 .. 47
  const int mb = mt * 16, nb = nt * 64;
  const int g = lane >> 4, l16 = lane & 15;

  const __bf16* arow  = xb  + (size_t)(mb + l16) * CC;
  const __bf16* brow0 = wbt + (size_t)(nb + l16) * CC;

  v8f acc[4];
#pragma unroll
  for (int j = 0; j < 4; ++j) acc[j] = zero8();

  for (int k0 = 0; k0 < CC; k0 += 32) {
    __builtin_prefetch(arow + k0 + 256, 0, 1);
    v16bf a = frag_rowA(arow, k0, g);
#pragma unroll
    for (int j = 0; j < 4; ++j) {
      v16bf b = frag_rowB(brow0 + (size_t)j * 16 * CC, k0, g);
      acc[j] = WMMA_BF16(a, b, acc[j]);
    }
  }

  const int region = nb / CC;          // 0=q 1=k 2=v
  const int h = (nb % CC) / DD;        // head

  if (region < 2) {
    __bf16* dst = (region == 0) ? qh : kh;
#pragma unroll
    for (int j = 0; j < 2; ++j) {
#pragma unroll
      for (int v = 0; v < 8; ++v) {
        int mrow = mb + v + 8 * g;
        int bb = mrow >> 11;           // / T
        int t  = mrow & (TT - 1);
        int d1 = j * 16 + l16, d2 = d1 + 32;
        float x1 = acc[j][v], x2 = acc[j + 2][v];
        float r1 = rope[t * DD + d1], r2 = rope[t * DD + d2];
        float s1, c1, s2, c2;
        __sincosf(r1, &s1, &c1);
        __sincosf(r2, &s2, &c2);
        float o1 = x1 * c1 - x2 * s1;  // x*cos + (-x2)*sin
        float o2 = x2 * c2 + x1 * s2;  // x*cos + ( x1)*sin
        size_t base = ((size_t)(bb * HH + h) * TT + t) * DD;
        dst[base + d1] = (__bf16)o1;
        dst[base + d2] = (__bf16)o2;
      }
    }
  } else {
#pragma unroll
    for (int j = 0; j < 4; ++j) {
#pragma unroll
      for (int v = 0; v < 8; ++v) {
        int mrow = mb + v + 8 * g;
        int bb = mrow >> 11;
        int t  = mrow & (TT - 1);
        int d  = j * 16 + l16;
        vt[((size_t)(bb * HH + h) * DD + d) * TT + t] = (__bf16)acc[j][v];
      }
    }
  }
}

// ---------------------------------------------------------------------------
// Flash attention: one wave per 16-query tile per (b,h).  No LDS, no barriers.
// S^T = K_tile @ Q   (lane holds fixed query -> softmax stats need only
// shfl_xor(16) across the lane pair).  O^T accumulated as Vt @ P^T where the
// P^T B-fragment is built from the S^T accumulators with 16 lane-pair shuffles.
__global__ __launch_bounds__(128) void attn_kernel(
    const __bf16* __restrict__ qh, const __bf16* __restrict__ kh,
    const __bf16* __restrict__ vt, const float* __restrict__ gval,
    __bf16* __restrict__ ybf) {
  const int lane = threadIdx.x & 31;
  const int w = blockIdx.x * 4 + (threadIdx.x >> 5);  // 0 .. 8191
  const int bh = w >> 7;                              // / (T/16)
  const int qt = w & 127;
  const int qb = qt * 16;
  const int b = bh >> 4, h = bh & 15;
  const int g = lane >> 4, l16 = lane & 15;
  const int q = qb + l16;                             // this lane's query row

  const __bf16* qrow  = qh + ((size_t)bh * TT + q) * DD;
  const __bf16* kbase = kh + (size_t)bh * TT * DD;
  const __bf16* vbase = vt + (size_t)bh * DD * TT;

  const v16bf q0 = frag_rowB(qrow, 0, g);
  const v16bf q1 = frag_rowB(qrow, 32, g);
  const float gq = 0.1f * gval[((size_t)(b * TT) + q) * HH + h];
  const float scale = 0.125f;  // D^-0.5

  float m = -1e30f, l = 0.f;
  v8f o[4];
#pragma unroll
  for (int j = 0; j < 4; ++j) o[j] = zero8();

  for (int kb = 0; kb < qb + 16; kb += 32) {
    int r0 = kb + l16;
    int r1 = kb + 16 + l16;
    if (r1 > TT - 1) r1 = TT - 1;  // fully-masked rows: clamp to stay in-buffer

    v16bf a00 = frag_rowA(kbase + (size_t)r0 * DD, 0, g);
    v16bf a01 = frag_rowA(kbase + (size_t)r0 * DD, 32, g);
    v16bf a10 = frag_rowA(kbase + (size_t)r1 * DD, 0, g);
    v16bf a11 = frag_rowA(kbase + (size_t)r1 * DD, 32, g);

    v8f s0 = zero8();
    s0 = WMMA_BF16(a00, q0, s0);
    s0 = WMMA_BF16(a01, q1, s0);
    v8f s1 = zero8();
    s1 = WMMA_BF16(a10, q0, s1);
    s1 = WMMA_BF16(a11, q1, s1);

    float att0[8], att1[8];
    float mx = -1e30f;
#pragma unroll
    for (int v = 0; v < 8; ++v) {
      int key0 = kb + v + 8 * g;
      int key1 = key0 + 16;
      float d0 = fminf((float)(q - key0), 64.f);
      float d1 = fminf((float)(q - key1), 64.f);
      float a0 = s0[v] * scale - gq * d0;
      float a1 = s1[v] * scale - gq * d1;
      if (key0 > q) a0 = -1e30f;
      if (key1 > q) a1 = -1e30f;
      att0[v] = a0;
      att1[v] = a1;
      mx = fmaxf(mx, fmaxf(a0, a1));
    }
    mx = fmaxf(mx, __shfl_xor(mx, 16, 32));
    float mnew = fmaxf(m, mx);
    float corr = __expf(m - mnew);

    float p0[8], p1[8];
    float rs = 0.f;
#pragma unroll
    for (int v = 0; v < 8; ++v) {
      p0[v] = __expf(att0[v] - mnew);
      p1[v] = __expf(att1[v] - mnew);
      rs += p0[v] + p1[v];
    }
    rs += __shfl_xor(rs, 16, 32);
    l = l * corr + rs;
    m = mnew;
#pragma unroll
    for (int j = 0; j < 4; ++j) o[j] = o[j] * corr;

    // Build P^T B-fragment: lane needs keys [16g, 16g+16) for its query.
    FragAB pf;
#pragma unroll
    for (int v = 0; v < 8; ++v) {
      float t0 = __shfl_xor(p0[v], 16, 32);  // partner half of first 16 keys
      float t1 = __shfl_xor(p1[v], 16, 32);  // partner half of second 16 keys
      float lo = (g == 0) ? p0[v] : t1;
      float hi = (g == 0) ? t0 : p1[v];
      pf.v[v] = (__bf16)lo;
      pf.v[v + 8] = (__bf16)hi;
    }

#pragma unroll
    for (int j = 0; j < 4; ++j) {
      v16bf vf = frag_rowA(vbase + (size_t)(j * 16 + l16) * TT, kb, g);
      o[j] = WMMA_BF16(vf, pf.v, o[j]);
    }
  }

  const float inv = 1.f / l;
  const size_t outrow = ((size_t)(b * TT) + q) * CC + h * DD;
#pragma unroll
  for (int j = 0; j < 4; ++j) {
#pragma unroll
    for (int v = 0; v < 8; ++v) {
      int d = j * 16 + v + 8 * g;
      ybf[outrow + d] = (__bf16)(o[j][v] * inv);
    }
  }
}

// ---------------------------------------------------------------------------
// out = ybf @ W_proj  (bf16 WMMA, fp32 out)
__global__ __launch_bounds__(128) void proj_gemm_kernel(
    const __bf16* __restrict__ ybf, const __bf16* __restrict__ wpt,
    float* __restrict__ out) {
  const int lane = threadIdx.x & 31;
  const int w = blockIdx.x * 4 + (threadIdx.x >> 5);  // 0 .. 8191
  const int mt = w % (MM / 16);
  const int nt = w / (MM / 16);                       // 0 .. 15
  const int mb = mt * 16, nb = nt * 64;
  const int g = lane >> 4, l16 = lane & 15;

  const __bf16* arow  = ybf + (size_t)(mb + l16) * CC;
  const __bf16* brow0 = wpt + (size_t)(nb + l16) * CC;

  v8f acc[4];
#pragma unroll
  for (int j = 0; j < 4; ++j) acc[j] = zero8();

  for (int k0 = 0; k0 < CC; k0 += 32) {
    __builtin_prefetch(arow + k0 + 256, 0, 1);
    v16bf a = frag_rowA(arow, k0, g);
#pragma unroll
    for (int j = 0; j < 4; ++j) {
      v16bf b = frag_rowB(brow0 + (size_t)j * 16 * CC, k0, g);
      acc[j] = WMMA_BF16(a, b, acc[j]);
    }
  }

#pragma unroll
  for (int j = 0; j < 4; ++j) {
#pragma unroll
    for (int v = 0; v < 8; ++v) {
      int mrow = mb + v + 8 * g;
      out[(size_t)mrow * CC + nb + j * 16 + l16] = acc[j][v];
    }
  }
}

// ---------------------------------------------------------------------------
extern "C" void kernel_launch(void* const* d_in, const int* in_sizes, int n_in,
                              void* d_out, int out_size, void* d_ws, size_t ws_size,
                              hipStream_t stream) {
  (void)in_sizes; (void)n_in; (void)out_size; (void)ws_size;
  const float* x      = (const float*)d_in[0];
  const float* rope   = (const float*)d_in[1];
  const float* W_attn = (const float*)d_in[2];
  const float* W_proj = (const float*)d_in[3];
  const float* g_w    = (const float*)d_in[4];
  const float* g_b    = (const float*)d_in[5];
  float* out = (float*)d_out;

  // workspace layout (bytes), all 256-aligned; total ~88.5 MiB
  char* ws = (char*)d_ws;
  size_t off = 0;
  __bf16* xb  = (__bf16*)(ws + off); off += (size_t)MM * CC * 2;   // 16 MiB
  __bf16* wbt = (__bf16*)(ws + off); off += (size_t)N3 * CC * 2;   //  6 MiB
  __bf16* wpt = (__bf16*)(ws + off); off += (size_t)CC * CC * 2;   //  2 MiB
  float*  gv  = (float*) (ws + off); off += (size_t)MM * HH * 4;   // .5 MiB
  __bf16* qh  = (__bf16*)(ws + off); off += (size_t)MM * CC * 2;   // 16 MiB
  __bf16* kh  = (__bf16*)(ws + off); off += (size_t)MM * CC * 2;   // 16 MiB
  __bf16* vt  = (__bf16*)(ws + off); off += (size_t)MM * CC * 2;   // 16 MiB
  __bf16* ybf = (__bf16*)(ws + off); off += (size_t)MM * CC * 2;   // 16 MiB

  // 1) precision conversion + weight transposes
  cvt_x_kernel<<<(MM * CC + 255) / 256, 256, 0, stream>>>(x, xb, MM * CC);
  cvt_wt_kernel<<<(CC * N3 + 255) / 256, 256, 0, stream>>>(W_attn, wbt, CC, N3);
  cvt_wt_kernel<<<(CC * CC + 255) / 256, 256, 0, stream>>>(W_proj, wpt, CC, CC);

  // 2) gravity gate: softplus(x @ g_w + g_b)
  gravity_kernel<<<MM, 256, 0, stream>>>(x, g_w, g_b, gv);

  // 3) qkv projection (WMMA) with fused RoPE + head reorg
  qkv_gemm_rope_kernel<<<(MM / 16) * (N3 / 64) / 4, 128, 0, stream>>>(
      xb, wbt, rope, qh, kh, vt);

  // 4) causal flash attention with gravity bias (WMMA)
  attn_kernel<<<(BB * HH) * (TT / 16) / 4, 128, 0, stream>>>(qh, kh, vt, gv, ybf);

  // 5) output projection (WMMA)
  proj_gemm_kernel<<<(MM / 16) * (CC / 64) / 4, 128, 0, stream>>>(ybf, wpt, out);
}
